// NNUE_71141838291202
// MI455X (gfx1250) — matrix operand order repaired and checked by reading, not compiled, not run on hardware
//
#include <hip/hip_runtime.h>

#define NF    40960
#define FTO   257
#define NPAD  272
#define BATCH 4096
#define KSPLIT 4
#define KCHUNK (NF / KSPLIT)   // 10240

#ifndef __has_builtin
#define __has_builtin(x) 0
#endif
#if __has_builtin(__builtin_amdgcn_global_load_async_to_lds_b128)
#define USE_ASYNC 1
#else
#define USE_ASYNC 0
#endif

typedef __attribute__((ext_vector_type(16))) _Float16 v16h;
typedef __attribute__((ext_vector_type(8)))  _Float16 h8;
typedef __attribute__((ext_vector_type(4)))  _Float16 h4;
typedef __attribute__((ext_vector_type(8)))  float    v8f;
typedef __attribute__((ext_vector_type(2)))  __fp16   g2;   // cvt_pkrtz result type
typedef __attribute__((ext_vector_type(4)))  __fp16   g4;

static __device__ __forceinline__ v8f wmma16(v16h a, v16h b, v8f c) {
  // D = A(16x32 f16) * B(32x16 f16) + C(16x16 f32)
  return __builtin_amdgcn_wmma_f32_16x16x32_f16(false, a, false, b, (short)0, c,
                                                false, false);
}

static __device__ __forceinline__ h4 pack4(float4 v) {
  g2 a = __builtin_amdgcn_cvt_pkrtz(v.x, v.y);
  g2 b = __builtin_amdgcn_cvt_pkrtz(v.z, v.w);
  g4 t = __builtin_shufflevector(a, b, 0, 1, 2, 3);
  return __builtin_bit_cast(h4, t);
}

static __device__ __forceinline__ v16h cat16(h8 lo, h8 hi) {
  return __builtin_shufflevector(lo, hi, 0, 1, 2, 3, 4, 5, 6, 7,
                                 8, 9, 10, 11, 12, 13, 14, 15);
}

// 16-byte copy global -> LDS. Async (ASYNCcnt, no VGPR payload) when the
// gfx1250 builtin exists; synchronous fallback otherwise.
#if USE_ASYNC
typedef int v4i __attribute__((vector_size(4 * sizeof(int))));
typedef __attribute__((address_space(1))) v4i* gv4p;   // global int4*
typedef __attribute__((address_space(3))) v4i* lv4p;   // LDS int4*
static __device__ __forceinline__ void cp16(const void* g, void* l) {
  // Generic->AS1 is value-preserving; the low 32 bits of a generic LDS
  // pointer are the wave-relative LDS byte offset (AS3 value).
  __builtin_amdgcn_global_load_async_to_lds_b128(
      (gv4p)(unsigned long long)g,
      (lv4p)(unsigned)(unsigned long long)l, 0, 0);
}
#else
static __device__ __forceinline__ void cp16(const void* g, void* l) {
  *(uint4*)l = *(const uint4*)g;
}
#endif

// Build a B operand (32x16 f16) from row-major f32 weights laid out [n][k]:
// this lane needs 16 consecutive K values starting at p.
static __device__ __forceinline__ v16h loadB16(const float* __restrict__ p) {
  v16h r;
#pragma unroll
  for (int q = 0; q < 4; ++q) {
    float4 v = *(const float4*)(p + 4 * q);
    h4 h = pack4(v);
    r[4 * q + 0] = h[0]; r[4 * q + 1] = h[1];
    r[4 * q + 2] = h[2]; r[4 * q + 3] = h[3];
  }
  return r;
}

static __device__ __forceinline__ float crelu(float x) {
  return fminf(fmaxf(x, 0.f), 1.f);
}

// ---------------------------------------------------------------------------
// Kernel 0: pre-convert ft_w (257 x 40960 f32) -> f16 [n][k], zero-padded to
// 320 rows so the GEMM staging needs no bounds checks. ~63 MB, one shot.
// ---------------------------------------------------------------------------
__global__ __launch_bounds__(256) void cvt_ftw_kernel(
    const float* __restrict__ ftw, _Float16* __restrict__ fB) {
  int idx = blockIdx.x * 256 + threadIdx.x;   // 0 .. 320*40960/4 - 1
  int r = idx / (NF / 4);
  int c = (idx % (NF / 4)) * 4;
  float4 v = make_float4(0.f, 0.f, 0.f, 0.f);
  if (r < FTO) v = *(const float4*)(ftw + (size_t)r * NF + c);
  *(h4*)(fB + (size_t)r * NF + c) = pack4(v);
}

// ---------------------------------------------------------------------------
// Kernel 1: P[ks][mat] = F[mat][:, kchunk] @ ft_w[:, kchunk].T  (no bias)
// Block: 256 threads = 8 waves; 32 M-rows x all 257 N-cols per block, so the
// 1.34 GB of wfts/bfts streams from HBM exactly once (the roofline floor).
// B tiles flow global->LDS via async copies (double buffered, one barrier +
// one s_wait_asynccnt per k-step); A tile keeps a 4-VGPR register prefetch
// for the f32->f16 convert. Branch-free WMMA inner loop (20 zero-padded
// N tiles, 5 per N-wave).
// ---------------------------------------------------------------------------
__global__ __launch_bounds__(256) void ft_gemm_kernel(
    const float* __restrict__ wfts, const float* __restrict__ bfts,
    const _Float16* __restrict__ fB, float* __restrict__ P) {
  __shared__ _Float16 sA[2][32 * 40];    // 32 rows x 32 K (stride 40)
  __shared__ _Float16 sB[2][320 * 40];   // 320 n-rows x 32 K (stride 40)

  const int tid = threadIdx.x;
  const int lane = tid & 31;
  const int wave = tid >> 5;
  const int mw = wave & 1;    // M sub-tile
  const int nw = wave >> 1;   // N wave: 5 tiles each, 20 total (17 valid)
  const int mrow = lane & 15;
  const int hi = lane >> 4;

  const int m0 = blockIdx.x * 32;
  const int mat = blockIdx.y;
  const int ks = blockIdx.z;
  const float* __restrict__ F = mat ? bfts : wfts;
  float* __restrict__ Pout = P + (size_t)(ks * 2 + mat) * BATCH * NPAD;
  const int kbeg = ks * KCHUNK;
  const int kend = kbeg + KCHUNK;

  v8f zero = {};
  v8f c[5];
#pragma unroll
  for (int t = 0; t < 5; ++t) c[t] = zero;

  // Staging geometry (per thread, fixed across k-steps)
  const int ar = tid >> 3;            // A row 0..31
  const int ak = (tid & 7) << 2;      // A k-offset (floats)
  const int br = tid >> 2;            // B row base 0..63 (+64*s)
  const int bc = (tid & 3) << 3;      // B k-offset (halves): 0,8,16,24

#define STAGE_B(BUFP, K0)                                                     \
  do {                                                                        \
    _Pragma("unroll") for (int s = 0; s < 5; ++s) {                           \
      cp16(fB + (size_t)(br + 64 * s) * NF + (K0) + bc,                       \
           (BUFP) + (br + 64 * s) * 40 + bc);                                 \
    }                                                                         \
  } while (0)

  float4 aReg = *(const float4*)(F + (size_t)(m0 + ar) * NF + kbeg + ak);
  STAGE_B(sB[0], kbeg);

  int buf = 0;
  for (int k0 = kbeg; k0 < kend; k0 += 32) {
    // Commit prefetched A to LDS[buf]; prefetch next A (wrap: harmless).
    *(h4*)(sA[buf] + ar * 40 + ak) = pack4(aReg);
    int knext = (k0 + 32 < kend) ? (k0 + 32) : kbeg;
    aReg = *(const float4*)(F + (size_t)(m0 + ar) * NF + knext + ak);

#if USE_ASYNC
    asm volatile("s_wait_asynccnt 0x0" ::: "memory");  // this buf's B landed
#endif
    __syncthreads();

    // Safe to start filling the other buffer: all reads of it finished
    // before every wave passed the barrier above.
    STAGE_B(sB[buf ^ 1], knext);

    // A operand: halves 0..15 -> K = hi*16+0..15
    const _Float16* ap = sA[buf] + (mw * 16 + mrow) * 40 + hi * 16;
    v16h av = cat16(*(const h8*)ap, *(const h8*)(ap + 8));
#pragma unroll
    for (int tt = 0; tt < 5; ++tt) {
      const _Float16* bp = sB[buf] + ((nw * 5 + tt) * 16 + mrow) * 40 + hi * 16;
      v16h bv = cat16(*(const h8*)bp, *(const h8*)(bp + 8));
      c[tt] = wmma16(av, bv, c[tt]);
    }
    buf ^= 1;
  }
#undef STAGE_B

  // D layout: lane<16 -> M=v, N=lane ; lane>=16 -> M=v+8, N=lane-16
#pragma unroll
  for (int tt = 0; tt < 5; ++tt) {
    int n = (nw * 5 + tt) * 16 + mrow;
    if (n < FTO) {
      int mb = m0 + mw * 16 + hi * 8;
#pragma unroll
      for (int v = 0; v < 8; ++v)
        Pout[(size_t)(mb + v) * NPAD + n] = c[tt][v];
    }
  }
}

// ---------------------------------------------------------------------------
// Kernel 2: sum 4 K-partials + ft_b, stm mixing, crelu, l1/l2 via WMMA,
// l3 + psqt epilogue. Block: 128 threads = 4 waves; 16 batch rows per wave.
// ---------------------------------------------------------------------------
__global__ __launch_bounds__(128) void nnue_head_kernel(
    const float* __restrict__ P, const float* __restrict__ stm,
    const float* __restrict__ ftb, const float* __restrict__ l1w,
    const float* __restrict__ l1b, const float* __restrict__ l2w,
    const float* __restrict__ l2b, const float* __restrict__ l3w,
    const float* __restrict__ l3b, float* __restrict__ out) {
  __shared__ _Float16 act[4][16 * 136];  // 128-wide K chunk of activations
  __shared__ _Float16 a2[4][16 * 40];    // 16x32 intermediate (A-operand form)
  __shared__ float ps[4][16];

  const int tid = threadIdx.x;
  const int lane = tid & 31;
  const int wave = tid >> 5;
  const int mrow = lane & 15;
  const int hi = lane >> 4;
  const int m0 = blockIdx.x * 64 + wave * 16;
  const size_t S = (size_t)BATCH * NPAD;

  if (lane < 16) {
    int m = m0 + lane;
    float s = stm[m];
    float wq = ftb[256], bq = ftb[256];
#pragma unroll
    for (int k = 0; k < KSPLIT; ++k) {
      wq += P[(k * 2 + 0) * S + (size_t)m * NPAD + 256];
      bq += P[(k * 2 + 1) * S + (size_t)m * NPAD + 256];
    }
    ps[wave][lane] = (wq + bq) * (s - 0.5f);
  }

  v8f c0 = {}, c1 = {};

  for (int kc = 0; kc < 4; ++kc) {  // four 128-wide K chunks of the 512 acc
    __syncthreads();
    // Build crelu(stm-mixed accumulator) chunk into LDS f16.
#pragma unroll
    for (int s8 = 0; s8 < 16; ++s8) {
      int idx = lane + s8 * 32;           // 0..511
      int r = idx >> 5;                   // row 0..15
      int q = idx & 31;                   // float4 within 128 cols
      int j = kc * 128 + q * 4;           // global col 0..508
      int m = m0 + r;
      float sv = stm[m];
      size_t rowoff = (size_t)m * NPAD;
      int jj = (j < 256) ? j : (j - 256);
      int wsel = (j < 256) ? 0 : 1;       // which partial set feeds va
      float4 va = make_float4(0.f, 0.f, 0.f, 0.f);
      float4 vb = make_float4(0.f, 0.f, 0.f, 0.f);
#pragma unroll
      for (int k = 0; k < KSPLIT; ++k) {
        float4 x = *(const float4*)(P + (k * 2 + wsel) * S + rowoff + jj);
        float4 y = *(const float4*)(P + (k * 2 + (wsel ^ 1)) * S + rowoff + jj);
        va.x += x.x; va.y += x.y; va.z += x.z; va.w += x.w;
        vb.x += y.x; vb.y += y.y; vb.z += y.z; vb.w += y.w;
      }
      float4 bias = *(const float4*)(ftb + jj);
      float4 o;
      o.x = crelu(sv * va.x + (1.f - sv) * vb.x + bias.x);
      o.y = crelu(sv * va.y + (1.f - sv) * vb.y + bias.y);
      o.z = crelu(sv * va.z + (1.f - sv) * vb.z + bias.z);
      o.w = crelu(sv * va.w + (1.f - sv) * vb.w + bias.w);
      *(h4*)(&act[wave][r * 136 + q * 4]) = pack4(o);
    }
    __syncthreads();

    // l1 GEMM over this chunk: 4 K-steps of 32, two 16-wide N tiles.
#pragma unroll
    for (int ksx = 0; ksx < 4; ++ksx) {
      const _Float16* ap = &act[wave][mrow * 136 + ksx * 32 + hi * 16];
      v16h av = cat16(*(const h8*)ap, *(const h8*)(ap + 8));
      int kg = kc * 128 + ksx * 32 + hi * 16;
      v16h b0 = loadB16(l1w + (size_t)mrow * 512 + kg);
      c0 = wmma16(av, b0, c0);
      v16h b1 = loadB16(l1w + (size_t)(16 + mrow) * 512 + kg);
      c1 = wmma16(av, b1, c1);
    }
  }

  __syncthreads();
  {  // l1 bias + crelu -> a2 in A-operand [m][k] layout (stride 40)
    float b0 = l1b[mrow], b1 = l1b[16 + mrow];
#pragma unroll
    for (int v = 0; v < 8; ++v) {
      int m = hi * 8 + v;
      a2[wave][m * 40 + mrow] = (_Float16)crelu(c0[v] + b0);
      a2[wave][m * 40 + 16 + mrow] = (_Float16)crelu(c1[v] + b1);
    }
  }
  __syncthreads();

  // l2: single K=32 step, two N tiles.
  v8f d0 = {}, d1 = {};
  {
    const _Float16* ap = &a2[wave][mrow * 40 + hi * 16];
    v16h av = cat16(*(const h8*)ap, *(const h8*)(ap + 8));
    v16h b0 = loadB16(l2w + (size_t)mrow * 32 + hi * 16);
    d0 = wmma16(av, b0, d0);
    v16h b1 = loadB16(l2w + (size_t)(16 + mrow) * 32 + hi * 16);
    d1 = wmma16(av, b1, d1);
  }
  __syncthreads();
  {  // l2 bias + crelu -> a2 (reuse)
    float b0 = l2b[mrow], b1 = l2b[16 + mrow];
#pragma unroll
    for (int v = 0; v < 8; ++v) {
      int m = hi * 8 + v;
      a2[wave][m * 40 + mrow] = (_Float16)crelu(d0[v] + b0);
      a2[wave][m * 40 + 16 + mrow] = (_Float16)crelu(d1[v] + b1);
    }
  }
  __syncthreads();

  // l3 + psqt epilogue: 16 lanes, one row each.
  if (lane < 16) {
    float acc = l3b[0];
#pragma unroll
    for (int j = 0; j < 32; ++j)
      acc += (float)a2[wave][lane * 40 + j] * l3w[j];
    out[m0 + lane] = acc + ps[wave][lane];
  }
}

extern "C" void kernel_launch(void* const* d_in, const int* in_sizes, int n_in,
                              void* d_out, int out_size, void* d_ws,
                              size_t ws_size, hipStream_t stream) {
  (void)in_sizes; (void)n_in; (void)out_size; (void)ws_size;
  const float* wfts = (const float*)d_in[0];
  const float* bfts = (const float*)d_in[1];
  const float* stm  = (const float*)d_in[2];
  const float* ftw  = (const float*)d_in[3];
  const float* ftb  = (const float*)d_in[4];
  const float* l1w  = (const float*)d_in[5];
  const float* l1b  = (const float*)d_in[6];
  const float* l2w  = (const float*)d_in[7];
  const float* l2b  = (const float*)d_in[8];
  const float* l3w  = (const float*)d_in[9];
  const float* l3b  = (const float*)d_in[10];

  // Workspace: 8 K-partial buffers (4 splits x {w,b}) then f16 ft_w copy.
  const size_t S = (size_t)BATCH * NPAD;            // floats per partial
  float* P = (float*)d_ws;                          // 8*S floats = 35.7 MB
  _Float16* fB = (_Float16*)(P + 8 * S);            // 320*40960 f16 = 26.2 MB

  cvt_ftw_kernel<<<(320 * (NF / 4)) / 256, 256, 0, stream>>>(ftw, fB);
  ft_gemm_kernel<<<dim3(BATCH / 32, 2, KSPLIT), 256, 0, stream>>>(wfts, bfts,
                                                                  fB, P);
  nnue_head_kernel<<<BATCH / 64, 128, 0, stream>>>(P, stm, ftb, l1w, l1b, l2w,
                                                   l2b, l3w, l3b,
                                                   (float*)d_out);
}
